// CausalMultiheadSelfAttention_70007966925513
// MI455X (gfx1250) — matrix operand order repaired and checked by reading, not compiled, not run on hardware
//
#include <hip/hip_runtime.h>
#include <hip/hip_bf16.h>

#define SEQ    4096
#define DM     1024
#define NH     16
#define DK     64
#define E3     3072

typedef __attribute__((ext_vector_type(16))) __bf16 v16bf;
typedef __attribute__((ext_vector_type(8)))  __bf16 v8bf;
typedef __attribute__((ext_vector_type(8)))  float  v8f;
typedef __attribute__((ext_vector_type(4)))  int    v4i;
typedef __attribute__((ext_vector_type(4)))  unsigned uint4v;

// ---------- gfx1250 async copy to LDS (guarded; sync fallback) ----------
#if __has_builtin(__builtin_amdgcn_global_load_async_to_lds_b128)
#define HAVE_ASYNC 1
typedef __attribute__((address_space(1))) v4i* gptr4;
typedef __attribute__((address_space(3))) v4i* lptr4;
#endif

__device__ __forceinline__ void async_cp16(void* lds, const void* g) {
#ifdef HAVE_ASYNC
    __builtin_amdgcn_global_load_async_to_lds_b128(
        (gptr4)(void*)g, (lptr4)lds, 0, 0);
#else
    *(uint4v*)lds = *(const uint4v*)g;
#endif
}
__device__ __forceinline__ void wait_async_all() {
#ifdef HAVE_ASYNC
#if __has_builtin(__builtin_amdgcn_s_wait_asynccnt)
    __builtin_amdgcn_s_wait_asynccnt(0);
#else
    asm volatile("s_wait_asynccnt 0" ::: "memory");
#endif
#endif
}

// ---------- bf16 helpers (RNE) ----------
__device__ __forceinline__ __bf16 f2bf(float f) {
    unsigned u = __builtin_bit_cast(unsigned, f);
    unsigned r = u + 0x7fffu + ((u >> 16) & 1u);
    unsigned short h = (unsigned short)(r >> 16);
    return __builtin_bit_cast(__bf16, h);
}
__device__ __forceinline__ float bf2f(__bf16 b) {
    unsigned short h = __builtin_bit_cast(unsigned short, b);
    unsigned u = ((unsigned)h) << 16;
    return __builtin_bit_cast(float, u);
}
__device__ __forceinline__ v16bf pack16(v8bf lo, v8bf hi) {
    v16bf r;
#pragma unroll
    for (int j = 0; j < 8; ++j) { r[j] = lo[j]; r[j + 8] = hi[j]; }
    return r;
}
__device__ __forceinline__ v8f zero8() {
    v8f z;
#pragma unroll
    for (int j = 0; j < 8; ++j) z[j] = 0.f;
    return z;
}

// ---------- kernel 1: f32 -> bf16 ----------
__global__ void cvt_bf16_kernel(const float* __restrict__ src,
                                __bf16* __restrict__ dst, int n) {
    for (int i = blockIdx.x * blockDim.x + threadIdx.x; i < n;
         i += gridDim.x * blockDim.x)
        dst[i] = f2bf(src[i]);
}

// ---------- kernel 2/5: C[MxN] = A[MxK] * B[NxK]^T  (NT GEMM, bf16 in, f32 acc)
// 128 threads = 4 waves; block tile 64x64; double-buffered async LDS staging.
// LDS rows padded to 72 elems (144B) to spread the 64 banks; keeps 16B align.
template <bool BF16OUT>
__global__ __launch_bounds__(128) void gemm_nt_kernel(
    const __bf16* __restrict__ A, const __bf16* __restrict__ B, void* Cout,
    int M, int N, int Kd) {
    __shared__ __align__(16) __bf16 At[2][64][72];
    __shared__ __align__(16) __bf16 Bt[2][64][72];

    const int t    = threadIdx.x;
    const int lane = t & 31;
    const int w    = t >> 5;
    const int hl   = lane >> 4;
    const int cl   = lane & 15;
    const int m0   = blockIdx.y * 64;
    const int n0   = blockIdx.x * 64;

    const int kbase = hl * 8;    // A layout: lanes16-31 hold K=8..15 / 24..31
    const int koff  = kbase;     // B layout: same as A (K=8..15 / 24..31)

    v8f acc[4];
#pragma unroll
    for (int nf = 0; nf < 4; ++nf) acc[nf] = zero8();

    // stage one 64(rows)x64(k) tile of A and of B: 512 16B-chunks each,
    // 4 per thread per matrix
    auto stage = [&](int buf, int kb) {
#pragma unroll
        for (int j = 0; j < 4; ++j) {
            int chunk = j * 128 + t;        // 0..511
            int row = chunk >> 3, cc = chunk & 7;
            async_cp16(&At[buf][row][cc * 8],
                       A + (size_t)(m0 + row) * Kd + kb + cc * 8);
            async_cp16(&Bt[buf][row][cc * 8],
                       B + (size_t)(n0 + row) * Kd + kb + cc * 8);
        }
    };

    stage(0, 0);
    const int ktiles = Kd >> 6;
    for (int i = 0; i < ktiles; ++i) {
        wait_async_all();
        __syncthreads();
        if (i + 1 < ktiles) stage((i + 1) & 1, (i + 1) * 64);

        const __bf16* Ab = &At[i & 1][w * 16][0];
        const __bf16* Bb = &Bt[i & 1][0][0];
#pragma unroll
        for (int ks = 0; ks < 2; ++ks) {
            // load A fragment + ALL FOUR B fragments into live regs first,
            // then 4 independent WMMAs (distinct accumulators, no RAW chain)
            const __bf16* ap = Ab + cl * 72 + ks * 32 + kbase;
            v16bf a = pack16(*(const v8bf*)ap, *(const v8bf*)(ap + 16));
            v16bf bf[4];
#pragma unroll
            for (int nf = 0; nf < 4; ++nf)
                bf[nf] = pack16(*(const v8bf*)(Bb + (nf * 16 + cl) * 72 + ks * 32 + kbase), *(const v8bf*)(Bb + (nf * 16 + cl) * 72 + ks * 32 + kbase + 16));
#pragma unroll
            for (int nf = 0; nf < 4; ++nf)
                acc[nf] = __builtin_amdgcn_wmma_f32_16x16x32_bf16(
                    false, a, false, bf[nf], (short)0, acc[nf], false, false);
        }
        __syncthreads();
    }

#pragma unroll
    for (int nf = 0; nf < 4; ++nf)
#pragma unroll
        for (int r = 0; r < 8; ++r) {
            int row = m0 + w * 16 + r + 8 * hl;
            int col = n0 + nf * 16 + cl;
            if (BF16OUT)
                ((__bf16*)Cout)[(size_t)row * N + col] = f2bf(acc[nf][r]);
            else
                ((float*)Cout)[(size_t)row * N + col] = acc[nf][r];
        }
}

// ---------- kernel 3: RoPE + per-head reorg ----------
// qkv[s][3072] -> Q[h][s][64], K[h][s][64] (rotated), Vt[h][64][s]
__global__ void rope_reorg_kernel(const __bf16* __restrict__ qkv,
                                  const int* __restrict__ pos,
                                  __bf16* __restrict__ Qo,
                                  __bf16* __restrict__ Ko,
                                  __bf16* __restrict__ Vto) {
    const int total = NH * SEQ * (DK / 2);
    for (int idx = blockIdx.x * blockDim.x + threadIdx.x; idx < total;
         idx += gridDim.x * blockDim.x) {
        int i = idx & 31;
        int s = (idx >> 5) & (SEQ - 1);
        int h = idx >> 17;
        float p   = (float)pos[s];
        float inv = __expf(-0.28782313662425572f * (float)i);  // 1e4^(-2i/64)
        float sn, cs;
        __sincosf(p * inv, &sn, &cs);

        size_t base = (size_t)s * E3 + h * DK + 2 * i;
        float qe = bf2f(qkv[base]),      qo = bf2f(qkv[base + 1]);
        float ke = bf2f(qkv[base + DM]), ko = bf2f(qkv[base + DM + 1]);
        __bf16 ve = qkv[base + 2 * DM],  vo = qkv[base + 2 * DM + 1];

        size_t hs = ((size_t)h * SEQ + s) * DK + 2 * i;
        Qo[hs]     = f2bf(qe * cs - qo * sn);
        Qo[hs + 1] = f2bf(qe * sn + qo * cs);
        Ko[hs]     = f2bf(ke * cs - ko * sn);
        Ko[hs + 1] = f2bf(ke * sn + ko * cs);
        Vto[((size_t)h * DK + 2 * i) * SEQ + s]     = ve;
        Vto[((size_t)h * DK + 2 * i + 1) * SEQ + s] = vo;
    }
}

// ---------- kernel 4: causal flash attention ----------
// grid (SEQ/64, NH); 128 threads = 4 waves; wave w -> q rows [qt*64+w*16, +16)
// K and V^T tiles double-buffered through LDS via async copies.
__global__ __launch_bounds__(128) void attn_kernel(
    const __bf16* __restrict__ Q, const __bf16* __restrict__ K,
    const __bf16* __restrict__ Vt, __bf16* __restrict__ AO) {
    __shared__ __align__(16) __bf16 Kt[2][64][72];   // [key_local][d]
    __shared__ __align__(16) __bf16 Vti[2][64][72];  // [d_local][key_local]
    __shared__ __align__(16) __bf16 Plds[4][16][72];

    const int t    = threadIdx.x;
    const int lane = t & 31;
    const int w    = t >> 5;
    const int hl   = lane >> 4;
    const int cl   = lane & 15;
    const int qt   = blockIdx.x;
    const int h    = blockIdx.y;
    const int r0   = qt * 64 + w * 16;

    const __bf16* Qh = Q  + (size_t)h * SEQ * DK;
    const __bf16* Kh = K  + (size_t)h * SEQ * DK;
    const __bf16* Vh = Vt + (size_t)h * DK * SEQ;

    const int kbase = hl * 8, koff = hl * 16;

    auto stage_kv = [&](int buf, int key0) {
#pragma unroll
        for (int j = 0; j < 4; ++j) {
            int chunk = j * 128 + t;
            int row = chunk >> 3, cc = chunk & 7;
            async_cp16(&Kt[buf][row][cc * 8],
                       Kh + (size_t)(key0 + row) * DK + cc * 8);
            async_cp16(&Vti[buf][row][cc * 8],
                       Vh + (size_t)row * SEQ + key0 + cc * 8);
        }
    };

    // Q fragments in registers for the whole kernel: two 16x32 chunks of d
    v16bf qa[2];
#pragma unroll
    for (int c = 0; c < 2; ++c) {
        const __bf16* qp = Qh + (size_t)(r0 + cl) * DK + c * 32 + kbase;
        qa[c] = pack16(*(const v8bf*)qp, *(const v8bf*)(qp + 16));
    }

    const float NEG = -3.0e38f;
    float mrun[8], lrun[8];
    v8f o[4];
#pragma unroll
    for (int r = 0; r < 8; ++r) { mrun[r] = NEG; lrun[r] = 0.f; }
#pragma unroll
    for (int nf = 0; nf < 4; ++nf) o[nf] = zero8();

    const float scale = 0.125f;  // 1/sqrt(64)

    stage_kv(0, 0);
    for (int kt = 0; kt <= qt; ++kt) {
        const int key0 = kt * 64;
        const bool diag = (kt == qt);
        const int buf = kt & 1;

        wait_async_all();
        __syncthreads();
        if (kt < qt) stage_kv(buf ^ 1, (kt + 1) * 64);

        // ---- S = Q * K^T: per k-chunk, batch 4 B-frag loads then 4
        //      independent WMMAs on distinct accumulators ----
        v8f sfr[4];
#pragma unroll
        for (int nf = 0; nf < 4; ++nf) sfr[nf] = zero8();
#pragma unroll
        for (int c = 0; c < 2; ++c) {
            v16bf kb[4];
#pragma unroll
            for (int nf = 0; nf < 4; ++nf)
                kb[nf] = pack16(*(const v8bf*)(&Kt[buf][nf * 16 + cl][c * 32 + kbase]), *(const v8bf*)(&Kt[buf][nf * 16 + cl][c * 32 + kbase + 16]));
#pragma unroll
            for (int nf = 0; nf < 4; ++nf)
                sfr[nf] = __builtin_amdgcn_wmma_f32_16x16x32_bf16(
                    false, qa[c], false, kb[nf], (short)0, sfr[nf], false, false);
        }

        // ---- online softmax over this key tile ----
#pragma unroll
        for (int r = 0; r < 8; ++r) {
            const int qrow = r0 + r + 8 * hl;
            float mx = NEG;
#pragma unroll
            for (int nf = 0; nf < 4; ++nf) {
                float v = sfr[nf][r] * scale;
                if (diag && (key0 + nf * 16 + cl > qrow)) v = NEG;
                sfr[nf][r] = v;
                mx = fmaxf(mx, v);
            }
#pragma unroll
            for (int off = 1; off < 16; off <<= 1)
                mx = fmaxf(mx, __shfl_xor(mx, off, 32));
            float mnew = fmaxf(mrun[r], mx);
            float corr = __expf(mrun[r] - mnew);
            float rowsum = 0.f;
#pragma unroll
            for (int nf = 0; nf < 4; ++nf) {
                float pv = __expf(sfr[nf][r] - mnew);
                sfr[nf][r] = pv;
                rowsum += pv;
            }
#pragma unroll
            for (int off = 1; off < 16; off <<= 1)
                rowsum += __shfl_xor(rowsum, off, 32);
            lrun[r] = lrun[r] * corr + rowsum;
            mrun[r] = mnew;
#pragma unroll
            for (int nf = 0; nf < 4; ++nf) o[nf][r] *= corr;
        }

        // ---- transpose P (C-layout -> A-layout) through LDS ----
#pragma unroll
        for (int r = 0; r < 8; ++r)
#pragma unroll
            for (int nf = 0; nf < 4; ++nf)
                Plds[w][r + 8 * hl][nf * 16 + cl] = f2bf(sfr[nf][r]);
        __syncthreads();
        v16bf pa[2];
#pragma unroll
        for (int c = 0; c < 2; ++c) {
            const __bf16* pp = &Plds[w][cl][c * 32 + kbase];
            pa[c] = pack16(*(const v8bf*)pp, *(const v8bf*)(pp + 16));
        }

        // ---- O += P * V: same batched-loads-then-4-WMMA structure ----
#pragma unroll
        for (int c = 0; c < 2; ++c) {
            v16bf vb[4];
#pragma unroll
            for (int nf = 0; nf < 4; ++nf)
                vb[nf] = pack16(*(const v8bf*)(&Vti[buf][nf * 16 + cl][c * 32 + kbase]), *(const v8bf*)(&Vti[buf][nf * 16 + cl][c * 32 + kbase + 16]));
#pragma unroll
            for (int nf = 0; nf < 4; ++nf)
                o[nf] = __builtin_amdgcn_wmma_f32_16x16x32_bf16(
                    false, pa[c], false, vb[nf], (short)0, o[nf], false, false);
        }
    }

    // ---- epilogue: O /= l, write back to [s][1024] ----
#pragma unroll
    for (int nf = 0; nf < 4; ++nf)
#pragma unroll
        for (int r = 0; r < 8; ++r) {
            float val = o[nf][r] / lrun[r];
            AO[(size_t)(r0 + r + 8 * hl) * DM + h * DK + nf * 16 + cl] = f2bf(val);
        }
}

// ---------- launch ----------
extern "C" void kernel_launch(void* const* d_in, const int* in_sizes, int n_in,
                              void* d_out, int out_size, void* d_ws, size_t ws_size,
                              hipStream_t stream) {
    (void)in_sizes; (void)n_in; (void)out_size; (void)ws_size;
    const float* x    = (const float*)d_in[0];
    const float* Wqkv = (const float*)d_in[1];
    const float* Wo   = (const float*)d_in[2];
    const int*   pos  = (const int*)d_in[3];

    char* ws = (char*)d_ws;
    __bf16* xb    = (__bf16*)(ws);                       //  8 MB
    __bf16* wqkvb = (__bf16*)(ws + (8u  << 20));         //  6 MB
    __bf16* wob   = (__bf16*)(ws + (14u << 20));         //  2 MB
    __bf16* qkvb  = (__bf16*)(ws + (16u << 20));         // 24 MB
    __bf16* Qb    = (__bf16*)(ws + (40u << 20));         //  8 MB
    __bf16* Kb    = (__bf16*)(ws + (48u << 20));         //  8 MB
    __bf16* Vtb   = (__bf16*)(ws + (56u << 20));         //  8 MB
    __bf16* aob   = (__bf16*)(ws + (64u << 20));         //  8 MB

    cvt_bf16_kernel<<<4096, 256, 0, stream>>>(x, xb, SEQ * DM);
    cvt_bf16_kernel<<<4096, 256, 0, stream>>>(Wqkv, wqkvb, E3 * DM);
    cvt_bf16_kernel<<<2048, 256, 0, stream>>>(Wo, wob, DM * DM);

    // qkv = x @ Wqkv^T : M=4096, N=3072, K=1024
    gemm_nt_kernel<true><<<dim3(E3 / 64, SEQ / 64), 128, 0, stream>>>(
        xb, wqkvb, (void*)qkvb, SEQ, E3, DM);

    rope_reorg_kernel<<<8192, 256, 0, stream>>>(qkvb, pos, Qb, Kb, Vtb);

    attn_kernel<<<dim3(SEQ / 64, NH), 128, 0, stream>>>(Qb, Kb, Vtb, aob);

    // out = attn @ Wo^T : M=4096, N=1024, K=1024, f32 out
    gemm_nt_kernel<false><<<dim3(DM / 64, SEQ / 64), 128, 0, stream>>>(
        aob, wob, d_out, SEQ, DM, DM);
}
